// Codebook_54580444397734
// MI455X (gfx1250) — matrix operand order repaired and checked by reading, not compiled, not run on hardware
//
#include <hip/hip_runtime.h>
#include <hip/hip_bf16.h>

// VQ-VAE codebook quantization for MI455X (gfx1250, wave32).
//
// argmin_k ||z_n - e_k||^2 over N=32768 rows, K=8192 codes, D=32, done as a
// fused GEMM (z * codebook^T via v_wmma_f32_16x16x32_bf16, hi/lo bf16 error-
// compensated split => ~f32-exact dots) + row-wise argmin + gather + loss.
// Compute-bound (AI ~1900 FLOP/B). K loop is software-pipelined TWO deep:
//  - B operands register double-buffered (loads 1 iteration ahead)
//  - argmin consumption delayed 1 iteration behind its producing WMMAs
// and the schedule is pinned with sched_group_barrier to interleave
// (VMEM, WMMA, 9xVALU) so WMMA->VALU hazard slots are filled with real work.

typedef __bf16 bf16_t;
typedef __attribute__((ext_vector_type(16))) __bf16 v16bf;
typedef __attribute__((ext_vector_type(8)))  float  v8f;

#define K_CODES 8192
#define DIM 32
#define N_ROWS 32768
#define WAVES_PER_BLOCK 8
#define ROWS_PER_BLOCK (WAVES_PER_BLOCK * 16)   // 128
#define GRID_MAIN (N_ROWS / ROWS_PER_BLOCK)     // 256
#define FLT_BIG 3.402823466e38f

#define WMMA_BF16(A, B, C) \
  __builtin_amdgcn_wmma_f32_16x16x32_bf16(false, (A), false, (B), (short)0, (C), false, false)

// IGLP scheduling pins (best-effort; no-op if builtin unavailable).
// Masks: 0x002 = VALU, 0x008 = MFMA/WMMA, 0x020 = VMEM read.
#if defined(__has_builtin)
#if __has_builtin(__builtin_amdgcn_sched_group_barrier)
#define SCHED_GROUP(mask, n) __builtin_amdgcn_sched_group_barrier((mask), (n), 0)
#endif
#endif
#ifndef SCHED_GROUP
#define SCHED_GROUP(mask, n)
#endif

// ---------------------------------------------------------------------------
// Pass 0: codebook -> bf16 hi/lo split + exact f32 squared norms (into d_ws).
// ---------------------------------------------------------------------------
__global__ void vq_prep(const float* __restrict__ cb,
                        bf16_t* __restrict__ hi, bf16_t* __restrict__ lo,
                        float* __restrict__ nrm) {
  int k = blockIdx.x * blockDim.x + threadIdx.x;
  if (k >= K_CODES) return;
  const float* row = cb + (size_t)k * DIM;
  float n = 0.f;
#pragma unroll
  for (int d = 0; d < DIM; ++d) {
    float v = row[d];
    n += v * v;
    bf16_t h = (bf16_t)v;                 // RNE to bf16
    float  r = v - (float)h;              // exact residual
    hi[k * DIM + d] = h;
    lo[k * DIM + d] = (bf16_t)r;
  }
  nrm[k] = n;
}

// ---------------------------------------------------------------------------
// Pass 1: fused distance GEMM + argmin + gather + loss partials.
// One wave owns a 16-row z tile; loops over 8192 codes, 32 per iteration
// (two 16x16 WMMA N-tiles).
// ---------------------------------------------------------------------------
__global__ __launch_bounds__(256) void vq_main(
    const float*  __restrict__ z,
    const float*  __restrict__ cb,
    const bf16_t* __restrict__ cbhi,
    const bf16_t* __restrict__ cblo,
    const float*  __restrict__ nrm,
    float* __restrict__ out_zq,
    float* __restrict__ out_idx,
    float* __restrict__ partials) {
  __shared__ int   s_idx[WAVES_PER_BLOCK * 16];
  __shared__ float s_red[256];

  const int tid  = threadIdx.x;
  const int wave = tid >> 5;
  const int lane = tid & 31;
  const int half = lane >> 4;   // half-wave
  const int ln   = lane & 15;   // M index (A) / N index (B,D) within tile

  const int row_base = (blockIdx.x * WAVES_PER_BLOCK + wave) * 16;
  const int my_row   = row_base + ln;

  // --- A operand (z tile, 16x32 bf16) in hi/lo split form. ---
  // 16-bit A layout: lane m=ln, element e: K = h*16+e (matches B)
  const float* zr = z + (size_t)my_row * DIM;
  v16bf a_hi, a_lo;
#pragma unroll
  for (int e = 0; e < 16; ++e) {
    int d = half * 16 + e;
    float v = zr[d];
    bf16_t h = (bf16_t)v;
    a_hi[e] = h;
    a_lo[e] = (bf16_t)(v - (float)h);
  }

  float best[8];
  int   bidx[8];
#pragma unroll
  for (int r = 0; r < 8; ++r) { best[r] = FLT_BIG; bidx[r] = 0; }

  // B layout (32x16 bf16): lane n=ln, element e: K = h*16 + e -> one
  // contiguous 32B (32B-aligned) run of converted codebook row (code, dims
  // [half*16, half*16+16)).
  const bf16_t* bh_base = cbhi + ln * DIM + half * 16;
  const bf16_t* bl_base = cblo + ln * DIM + half * 16;

  // Prologue: load B for codes [0,32).
  v16bf c0h = *(const v16bf*)(bh_base);
  v16bf c0l = *(const v16bf*)(bl_base);
  v16bf c1h = *(const v16bf*)(bh_base + 16 * DIM);
  v16bf c1l = *(const v16bf*)(bl_base + 16 * DIM);
  float e0 = nrm[ln];
  float e1 = nrm[16 + ln];

  // Delayed-consumption state: previous iteration's accumulators/candidates.
  // pe = FLT_BIG makes the first (bogus) update evaluate to FLT_BIG, which
  // strict '<' against FLT_BIG-initialized best never selects.
  v8f pp0 = {}, qq0 = {}, pp1 = {}, qq1 = {};
  float pe0 = FLT_BIG, pe1 = FLT_BIG;
  int   pc0 = 0, pc1 = 0;

#pragma unroll 2
  for (int cb0 = 0; cb0 < K_CODES; cb0 += 32) {
    // Prefetch next iteration's B operands (clamped redundant reload on the
    // final iteration; values unused).
    int nx = (cb0 + 32 < K_CODES) ? (cb0 + 32) : cb0;
    v16bf n0h = *(const v16bf*)(bh_base + nx * DIM);
    v16bf n0l = *(const v16bf*)(bl_base + nx * DIM);
    v16bf n1h = *(const v16bf*)(bh_base + (nx + 16) * DIM);
    v16bf n1l = *(const v16bf*)(bl_base + (nx + 16) * DIM);
    float ne0 = nrm[nx + ln];
    float ne1 = nrm[nx + 16 + ln];

    // Four independent accumulator chains of length 2:
    //   p = a_hi*b_hi + a_lo*b_lo,  q = a_hi*b_lo + a_lo*b_hi, dot = p + q.
    v8f p0 = {}, q0 = {}, p1 = {}, q1 = {};
    p0 = WMMA_BF16(a_hi, c0h, p0);
    p1 = WMMA_BF16(a_hi, c1h, p1);
    q0 = WMMA_BF16(a_hi, c0l, q0);
    q1 = WMMA_BF16(a_hi, c1l, q1);
    p0 = WMMA_BF16(a_lo, c0l, p0);
    p1 = WMMA_BF16(a_lo, c1l, p1);
    q0 = WMMA_BF16(a_lo, c0h, q0);
    q1 = WMMA_BF16(a_lo, c1h, q1);

    // Consume PREVIOUS iteration's results (producers >= 8 WMMAs back).
    // D layout: lane holds column n=ln; VGPR r is row m = r + half*8.
    // dist (up to +||z||^2 const) = ||e||^2 - 2*(p+q). Strict '<' keeps the
    // first (lowest-index) occurrence, matching jnp.argmin.
#pragma unroll
    for (int r = 0; r < 8; ++r) {
      float d0 = __builtin_fmaf(-2.0f, qq0[r], __builtin_fmaf(-2.0f, pp0[r], pe0));
      if (d0 < best[r]) { best[r] = d0; bidx[r] = pc0; }
      float d1 = __builtin_fmaf(-2.0f, qq1[r], __builtin_fmaf(-2.0f, pp1[r], pe1));
      if (d1 < best[r]) { best[r] = d1; bidx[r] = pc1; }
    }

    // Rotate pipeline state.
    pp0 = p0; qq0 = q0; pp1 = p1; qq1 = q1;
    pe0 = e0; pe1 = e1;
    pc0 = cb0 + ln; pc1 = cb0 + 16 + ln;
    c0h = n0h; c0l = n0l; c1h = n1h; c1l = n1l;
    e0 = ne0; e1 = ne1;

    // Pin the per-iteration schedule: spread the 10 prefetch loads between
    // WMMAs and force >= 9 VALU (argmin work) after every WMMA so the 4-slot
    // WMMA->VALU hazard windows are filled with real instructions, not nops.
#pragma unroll
    for (int g = 0; g < 8; ++g) {
      SCHED_GROUP(0x020, 1);   // 1 VMEM read
      SCHED_GROUP(0x008, 1);   // 1 WMMA
      SCHED_GROUP(0x002, 9);   // 9 VALU
    }
    SCHED_GROUP(0x020, 2);     // remaining norm loads
  }

  // Epilogue: drain the last chunk's delayed update.
#pragma unroll
  for (int r = 0; r < 8; ++r) {
    float d0 = __builtin_fmaf(-2.0f, qq0[r], __builtin_fmaf(-2.0f, pp0[r], pe0));
    if (d0 < best[r]) { best[r] = d0; bidx[r] = pc0; }
    float d1 = __builtin_fmaf(-2.0f, qq1[r], __builtin_fmaf(-2.0f, pp1[r], pe1));
    if (d1 < best[r]) { best[r] = d1; bidx[r] = pc1; }
  }

  // --- argmin across the 16 N-lanes of each half-wave (xor masks < 16 stay
  // within a half). Tie-break: smaller index (argmin semantics). ---
#pragma unroll
  for (int r = 0; r < 8; ++r) {
    float d = best[r];
    int   i = bidx[r];
#pragma unroll
    for (int m = 8; m >= 1; m >>= 1) {
      float od = __shfl_xor(d, m, 32);
      int   oi = __shfl_xor(i, m, 32);
      if (od < d || (od == d && oi < i)) { d = od; i = oi; }
    }
    if (ln == 0) s_idx[wave * 16 + half * 8 + r] = i;  // row m = r + half*8
  }
  __syncthreads();

  // --- Gather z_q, write indices, accumulate squared error. ---
  float acc_loss = 0.f;
#pragma unroll 4
  for (int m = 0; m < 16; ++m) {
    int idx = s_idx[wave * 16 + m];
    int row = row_base + m;
    float zq = cb[idx * DIM + lane];           // 32 lanes cover 32 dims
    float zv = z[(size_t)row * DIM + lane];
    out_zq[(size_t)row * DIM + lane] = zq;     // z_q_st == z_q numerically
    float df = zq - zv;
    acc_loss += df * df;
    if (lane == 0) out_idx[row] = (float)idx;
  }

  // Fixed-order block tree reduction -> per-block partial (no float atomics;
  // keeps kernel_launch bit-deterministic across graph replays).
  s_red[tid] = acc_loss;
  __syncthreads();
#pragma unroll
  for (int s = 128; s > 0; s >>= 1) {
    if (tid < s) s_red[tid] += s_red[tid + s];
    __syncthreads();
  }
  if (tid == 0) partials[blockIdx.x] = s_red[0];
}

// ---------------------------------------------------------------------------
// Pass 2: deterministic final reduction of 256 partials -> loss.
// loss = (1 + BETA) * MSE = 1.25 * mean((zq - z)^2)  (stop_gradient does not
// change forward values; z_q_st == z_q).
// ---------------------------------------------------------------------------
__global__ void vq_finalize(const float* __restrict__ partials,
                            float* __restrict__ loss) {
  __shared__ float s[256];
  int tid = threadIdx.x;
  s[tid] = partials[tid];
  __syncthreads();
#pragma unroll
  for (int st = 128; st > 0; st >>= 1) {
    if (tid < st) s[tid] += s[tid + st];
    __syncthreads();
  }
  if (tid == 0) loss[0] = s[0] * (1.25f / (float)((size_t)N_ROWS * DIM));
}

extern "C" void kernel_launch(void* const* d_in, const int* in_sizes, int n_in,
                              void* d_out, int out_size, void* d_ws, size_t ws_size,
                              hipStream_t stream) {
  const float* z  = (const float*)d_in[0];   // [8,4096,32] f32
  const float* cb = (const float*)d_in[1];   // [8192,32]   f32

  // Workspace layout (1,082,368 bytes total):
  char* ws = (char*)d_ws;
  bf16_t* cbhi     = (bf16_t*)(ws);                      // 512 KB
  bf16_t* cblo     = (bf16_t*)(ws + 524288);             // 512 KB
  float*  nrm      = (float*)(ws + 1048576);             //  32 KB
  float*  partials = (float*)(ws + 1048576 + 32768);     //   1 KB

  float* out      = (float*)d_out;
  float* out_zq   = out;                                  // 1048576 f32
  float* out_idx  = out + (size_t)N_ROWS * DIM;           //   32768 f32
  float* out_loss = out_idx + N_ROWS;                     //       1 f32

  vq_prep<<<K_CODES / 256, 256, 0, stream>>>(cb, cbhi, cblo, nrm);
  vq_main<<<GRID_MAIN, 256, 0, stream>>>(z, cb, cbhi, cblo, nrm,
                                         out_zq, out_idx, partials);
  vq_finalize<<<1, 256, 0, stream>>>(partials, out_loss);
}
